// HyperbolicRecurrentRGCN_35725537968815
// MI455X (gfx1250) — matrix operand order repaired and checked by reading, not compiled, not run on hardware
//
#include <hip/hip_runtime.h>
#include <math.h>

// ---------------- types / constants ----------------
typedef __attribute__((ext_vector_type(16))) __bf16 v16bf;
typedef __attribute__((ext_vector_type(8)))  float  v8f;

#define DD 256
#define SC 0.1f                  // sqrt(c), c = 0.01
#define EPSN 1e-5f
#define BALL_EPS 1e-3f
#define RRELU_SLOPE ((1.0f/8.0f + 1.0f/3.0f)*0.5f)

#define WAVES 4
#define ROWS_PER_WAVE 16
#define ROWS_PER_BLOCK (WAVES*ROWS_PER_WAVE)   // 64
#define KTBLK 8192               // elems per kt-block per matrix (32K x 256N bf16)

// ---------------- radial Poincare maps (scale factors) ----------------
__device__ __forceinline__ float logmap0_scale(float n) {
  n = fmaxf(n, EPSN);
  float scn = fminf(SC * n, 1.0f - BALL_EPS);
  return atanhf(scn) / (SC * n);
}
__device__ __forceinline__ float expmap0_scale(float n) {
  n = fmaxf(n, EPSN);
  float t = tanhf(SC * n);
  float s = t / (SC * n);
  float yn = fmaxf(t / SC, EPSN);
  float maxn = (1.0f - BALL_EPS) / SC;
  if (yn > maxn) s *= maxn / yn;
  return s;
}
__device__ __forceinline__ float explog0_scale(float n) {
  float se = expmap0_scale(n);
  float m = se * fmaxf(n, EPSN);
  return se * logmap0_scale(m);
}
__device__ __forceinline__ float sigmoidf_(float x) {
  return 1.0f / (1.0f + expf(-x));
}

// ---------------- weight prep: transpose + bf16 hi/lo split, WMMA-tile-major ----------------
// Logical B[k][j] per matrix:
//   mat 0: time_lin_w[j*2D + k] ; mat 1: w_neighbor[k*D+j] ; mat 2: w_loop ; mat 3: time_gate_w
// Packed layout: idx = ((kt*16 + nt)*32 + lane)*16 + e
//   kt=k>>5, nt=j>>4, lane=(j&15)|(((k>>4)&1)<<4), e=k&15
__global__ void prep_weights_kernel(const float* __restrict__ tlw,
                                    const float* __restrict__ wn,
                                    const float* __restrict__ wl,
                                    const float* __restrict__ wg,
                                    __bf16* __restrict__ hi,
                                    __bf16* __restrict__ lo) {
  int tid = blockIdx.x * blockDim.x + threadIdx.x;
  if (tid >= 4 * DD * DD) return;
  int m = tid >> 16;
  int idx = tid & 65535;
  int k = idx >> 8, j = idx & 255;
  float v;
  if      (m == 0) v = tlw[(size_t)j * (2 * DD) + k];
  else if (m == 1) v = wn[idx];
  else if (m == 2) v = wl[idx];
  else             v = wg[idx];
  int kt = k >> 5, nt = j >> 4;
  int lane = (j & 15) | (((k >> 4) & 1) << 4);
  int e = k & 15;
  size_t dst = (size_t)m * (DD * DD) + (((size_t)(kt * 16 + nt) * 32 + lane) << 4) + e;
  __bf16 h = (__bf16)v;
  hi[dst] = h;
  lo[dst] = (__bf16)(v - (float)h);
}

// ---------------- per-step time vector ----------------
__global__ void time_vec_kernel(const float* __restrict__ tw,
                                const float* __restrict__ tb,
                                const float* __restrict__ tlw,
                                const float* __restrict__ tlb,
                                float t_pos, float* __restrict__ tv) {
  __shared__ float te[DD];
  int j = threadIdx.x;
  te[j] = cosf(tw[j] * t_pos + tb[j]);
  __syncthreads();
  float acc = tlb[j];
  for (int k = 0; k < DD; ++k)
    acc += te[k] * tlw[(size_t)j * (2 * DD) + DD + k];
  tv[j] = acc;
}

// ---------------- async cooperative kt-block staging (global -> LDS) ----------------
// 128 threads move 16KB (hi) + 16KB (lo): 4 rounds x 32B x 2 = 8 async instructions/wave.
__device__ __forceinline__ void issue_ktblock(const __bf16* __restrict__ Bhi,
                                              const __bf16* __restrict__ Blo,
                                              int kt, unsigned lhi, unsigned llo, int tid) {
  const __bf16* shi = Bhi + (size_t)kt * KTBLK;
  const __bf16* slo = Blo + (size_t)kt * KTBLK;
#pragma unroll
  for (int c = 0; c < 4; ++c) {
    int off = (c * 128 + tid) * 16;            // elements (32B chunks)
    asm volatile("global_load_async_to_lds_b128 %0, %1, off"
                 :: "v"(lhi + (unsigned)off * 2), "v"(shi + off) : "memory");
    asm volatile("global_load_async_to_lds_b128 %0, %1, off"
                 :: "v"(llo + (unsigned)off * 2), "v"(slo + off) : "memory");
  }
}

// ---------------- A operand: 16x32 bf16 hi/lo from f32 LDS slab ----------------
// lane<16: row=lane, K = kt*32 + {0..15}; lane>=16: row=lane-16, K = kt*32 + {16..31}
__device__ __forceinline__ void load_atile(const float* __restrict__ xs, int kt, int lane,
                                           v16bf* ahi, v16bf* alo) {
  int m = lane & 15;
  int kbase = kt * 32 + ((lane & 16) ? 16 : 0);
  const float4* p4 = (const float4*)(xs + m * DD + kbase);
  float4 q0 = p4[0], q1 = p4[1], q2 = p4[2], q3 = p4[3];
  float f[16] = {q0.x, q0.y, q0.z, q0.w, q1.x, q1.y, q1.z, q1.w,
                 q2.x, q2.y, q2.z, q2.w, q3.x, q3.y, q3.z, q3.w};
#pragma unroll
  for (int j = 0; j < 16; ++j) {
    __bf16 h = (__bf16)f[j];
    (*ahi)[j] = h;
    (*alo)[j] = (__bf16)(f[j] - (float)h);
  }
}

// ---------------- cooperative, double-buffered, error-compensated bf16 GEMM ----------------
// acc[16] += xs(16x256) @ B(256x256); B streamed kt-block-wise via async-to-LDS.
// bh/bl: single LDS bases of 2*KTBLK elems; ping-pong by index arithmetic only (keeps AS3 -> ds ops).
__device__ __forceinline__ void gemm_slab_coop(const float* __restrict__ xs,
                                               const __bf16* __restrict__ Bhi,
                                               const __bf16* __restrict__ Blo,
                                               __bf16* __restrict__ bh,
                                               __bf16* __restrict__ bl,
                                               int lane, int tid, v8f acc[16]) {
  unsigned lhi = (unsigned)(uintptr_t)bh;
  unsigned llo = (unsigned)(uintptr_t)bl;
  issue_ktblock(Bhi, Blo, 0, lhi, llo, tid);
#pragma unroll 1
  for (int kt = 0; kt < 8; ++kt) {
    int cur = kt & 1;
    if (kt < 7) {
      issue_ktblock(Bhi, Blo, kt + 1,
                    lhi + (unsigned)(1 - cur) * (KTBLK * 2),
                    llo + (unsigned)(1 - cur) * (KTBLK * 2), tid);
      asm volatile("s_wait_asynccnt 0x8" ::: "memory");   // batch kt complete, kt+1 in flight
    } else {
      asm volatile("s_wait_asynccnt 0x0" ::: "memory");
    }
    __syncthreads();                                       // all waves' portions visible
    v16bf ahi, alo;
    load_atile(xs, kt, lane, &ahi, &alo);
    int bofs = cur * KTBLK + (lane << 4);
#pragma unroll
    for (int nt = 0; nt < 16; ++nt) {
      v16bf bhi = *(const v16bf*)(bh + bofs + nt * 512);
      v16bf blo = *(const v16bf*)(bl + bofs + nt * 512);
      acc[nt] = __builtin_amdgcn_wmma_f32_16x16x32_bf16(false, ahi, false, bhi,
                                                        (short)0, acc[nt], false, false);
      acc[nt] = __builtin_amdgcn_wmma_f32_16x16x32_bf16(false, ahi, false, blo,
                                                        (short)0, acc[nt], false, false);
      acc[nt] = __builtin_amdgcn_wmma_f32_16x16x32_bf16(false, alo, false, bhi,
                                                        (short)0, acc[nt], false, false);
    }
    __syncthreads();                                       // done reading buf[cur]
  }
}

// ---------------- accumulator -> global rows ----------------
__device__ __forceinline__ void store_acc_rows(float* __restrict__ outp, const v8f acc[16],
                                               int row0, int lane, int N) {
  int mbase = (lane & 16) ? 8 : 0;
  int colin = lane & 15;
#pragma unroll
  for (int nt = 0; nt < 16; ++nt) {
#pragma unroll
    for (int v = 0; v < 8; ++v) {
      int r = row0 + mbase + v;
      if (r < N) outp[(size_t)r * DD + nt * 16 + colin] = acc[nt][v];
    }
  }
}

// ---------------- row sum-of-squares: all 32 lanes, one half-row each ----------------
__device__ __forceinline__ float row_sumsq(const float (*xs)[DD], int lane) {
  int rr = lane & 15;
  int k0 = (lane & 16) ? (DD / 2) : 0;
  const float4* p = (const float4*)(&xs[rr][k0]);
  float ss = 0.f;
#pragma unroll
  for (int k = 0; k < DD / 8; ++k) {
    float4 v = p[k];
    ss += v.x * v.x + v.y * v.y + v.z * v.z + v.w * v.w;
  }
  ss += __shfl_xor(ss, 16, 32);
  return ss;                                   // full sum of row (lane&15), on every lane
}

// ---------------- Kernel A: temporal node update + projections ----------------
__global__ __launch_bounds__(WAVES * 32)
void node_update_kernel(const float* __restrict__ hin, const float* __restrict__ tv,
                        const __bf16* __restrict__ W1hi, const __bf16* __restrict__ W1lo,
                        const __bf16* __restrict__ WNhi, const __bf16* __restrict__ WNlo,
                        const __bf16* __restrict__ WLhi, const __bf16* __restrict__ WLlo,
                        float* __restrict__ ht, float* __restrict__ hW, float* __restrict__ hL,
                        int N, int first) {
  __shared__ float xs_all[WAVES][ROWS_PER_WAVE][DD];      // 64 KB
  __shared__ __align__(32) __bf16 bhbuf[2 * KTBLK];       // 32 KB
  __shared__ __align__(32) __bf16 blbuf[2 * KTBLK];       // 32 KB
  int tid = threadIdx.x;
  int lane = tid & 31;
  int wave = tid >> 5;
  float(*xs)[DD] = xs_all[wave];
  int row0 = blockIdx.x * ROWS_PER_BLOCK + wave * ROWS_PER_WAVE;
  int c0 = lane * 8;

  // Phase 1: load rows (float4 stripes)
  for (int i = 0; i < 16; ++i) {
    int r = row0 + i;
    float4 a = {0, 0, 0, 0}, b = {0, 0, 0, 0};
    if (r < N) {
      const float4* p = (const float4*)(hin + (size_t)r * DD + c0);
      a = p[0]; b = p[1];
    }
    float4* q = (float4*)(&xs[i][c0]);
    q[0] = a; q[1] = b;
  }
  __syncthreads();
  // radial map 1: logmap0(h) (first step: logmap0(expmap0(emb)))
  {
    float n = sqrtf(row_sumsq(xs, lane));
    float myscale = first ? explog0_scale(n) : logmap0_scale(n);
    for (int i = 0; i < 16; ++i) {
      float s = __shfl(myscale, i, 32);
#pragma unroll
      for (int j = 0; j < 8; ++j) xs[i][c0 + j] *= s;
    }
  }
  __syncthreads();

  // Phase 2: y = x @ W1T + tv -> ht = logmap0(expmap0(y))
  {
    v8f acc[16];
#pragma unroll
    for (int nt = 0; nt < 16; ++nt) acc[nt] = (v8f){0, 0, 0, 0, 0, 0, 0, 0};
    gemm_slab_coop(&xs[0][0], W1hi, W1lo, bhbuf, blbuf, lane, tid, acc);
    int mbase = (lane & 16) ? 8 : 0;
    int colin = lane & 15;
#pragma unroll
    for (int nt = 0; nt < 16; ++nt) {
      int col = nt * 16 + colin;
      float tvc = tv[col];
#pragma unroll
      for (int v = 0; v < 8; ++v) xs[mbase + v][col] = acc[nt][v] + tvc;
    }
  }
  __syncthreads();
  {
    float myscale = explog0_scale(sqrtf(row_sumsq(xs, lane)));
    for (int i = 0; i < 16; ++i) {
      float s = __shfl(myscale, i, 32);
#pragma unroll
      for (int j = 0; j < 8; ++j) xs[i][c0 + j] *= s;
    }
  }
  __syncthreads();
  // write ht (float4 stripes)
  for (int i = 0; i < 16; ++i) {
    int r = row0 + i;
    if (r < N) {
      const float4* q = (const float4*)(&xs[i][c0]);
      float4* p = (float4*)(ht + (size_t)r * DD + c0);
      p[0] = q[0]; p[1] = q[1];
    }
  }

  // Phase 3: hW = ht @ WN
  {
    v8f acc[16];
#pragma unroll
    for (int nt = 0; nt < 16; ++nt) acc[nt] = (v8f){0, 0, 0, 0, 0, 0, 0, 0};
    gemm_slab_coop(&xs[0][0], WNhi, WNlo, bhbuf, blbuf, lane, tid, acc);
    store_acc_rows(hW, acc, row0, lane, N);
  }
  // Phase 4: hL = ht @ WL
  {
    v8f acc[16];
#pragma unroll
    for (int nt = 0; nt < 16; ++nt) acc[nt] = (v8f){0, 0, 0, 0, 0, 0, 0, 0};
    gemm_slab_coop(&xs[0][0], WLhi, WLlo, bhbuf, blbuf, lane, tid, acc);
    store_acc_rows(hL, acc, row0, lane, N);
  }
}

// ---------------- edge scatter: relation sums ----------------
__global__ void rel_scatter_kernel(const int* __restrict__ src, const int* __restrict__ etype,
                                   const float* __restrict__ ht,
                                   float* __restrict__ rel_sum, float* __restrict__ rel_cnt,
                                   long long E) {
  long long tid = (long long)blockIdx.x * blockDim.x + threadIdx.x;
  if (tid >= E * 64) return;
  int i = (int)(tid >> 6);
  int c = (int)(tid & 63);
  int s = src[i], e = etype[i];
  float4 v = *(const float4*)(ht + (size_t)s * DD + c * 4);
  float* p = rel_sum + (size_t)e * DD + c * 4;
  atomicAdd(p + 0, v.x); atomicAdd(p + 1, v.y);
  atomicAdd(p + 2, v.z); atomicAdd(p + 3, v.w);
  if (c == 0) atomicAdd(rel_cnt + e, 1.0f);
}

// ---------------- GRU over relations + rW = r_emb_new @ w_neighbor ----------------
__global__ __launch_bounds__(256)
void gru_kernel(const float* __restrict__ emb_rel, const float* __restrict__ rel_sum,
                const float* __restrict__ rel_cnt,
                const float* __restrict__ w_ih, const float* __restrict__ w_hh,
                const float* __restrict__ b_ih, const float* __restrict__ b_hh,
                const float* __restrict__ w_neighbor,
                float* __restrict__ r_emb, float* __restrict__ rW) {
  __shared__ float x[2 * DD];
  __shared__ float h[DD];
  __shared__ float hn[DD];
  int r = blockIdx.x, j = threadIdx.x;
  float cnt = fmaxf(rel_cnt[r], 1.0f);
  x[j]      = emb_rel[(size_t)r * DD + j];
  x[DD + j] = rel_sum[(size_t)r * DD + j] / cnt;
  h[j]      = r_emb[(size_t)r * DD + j];
  __syncthreads();
  const float* wr = w_ih + (size_t)(0 * DD + j) * (2 * DD);
  const float* wz = w_ih + (size_t)(1 * DD + j) * (2 * DD);
  const float* wn = w_ih + (size_t)(2 * DD + j) * (2 * DD);
  float ir = b_ih[j], iz = b_ih[DD + j], in = b_ih[2 * DD + j];
  for (int k = 0; k < 2 * DD; ++k) {
    float xv = x[k];
    ir += xv * wr[k]; iz += xv * wz[k]; in += xv * wn[k];
  }
  const float* vr = w_hh + (size_t)(0 * DD + j) * DD;
  const float* vz = w_hh + (size_t)(1 * DD + j) * DD;
  const float* vn = w_hh + (size_t)(2 * DD + j) * DD;
  float hr = b_hh[j], hz = b_hh[DD + j], hnv = b_hh[2 * DD + j];
  for (int k = 0; k < DD; ++k) {
    float hv = h[k];
    hr += hv * vr[k]; hz += hv * vz[k]; hnv += hv * vn[k];
  }
  float rg = sigmoidf_(ir + hr);
  float zg = sigmoidf_(iz + hz);
  float ng = tanhf(in + rg * hnv);
  float out = (1.0f - zg) * ng + zg * h[j];
  hn[j] = out;
  r_emb[(size_t)r * DD + j] = out;
  __syncthreads();
  float acc = 0.f;
  for (int k = 0; k < DD; ++k) acc += hn[k] * w_neighbor[(size_t)k * DD + j];
  rW[(size_t)r * DD + j] = acc;
}

// ---------------- edge scatter: node aggregation (msg = hW[s] + rW[e]) ----------------
__global__ void agg_scatter_kernel(const int* __restrict__ src, const int* __restrict__ dst,
                                   const int* __restrict__ etype,
                                   const float* __restrict__ hW, const float* __restrict__ rW,
                                   float* __restrict__ agg, float* __restrict__ deg,
                                   long long E) {
  long long tid = (long long)blockIdx.x * blockDim.x + threadIdx.x;
  if (tid >= E * 64) return;
  int i = (int)(tid >> 6);
  int c = (int)(tid & 63);
  int s = src[i], d = dst[i], e = etype[i];
  float4 a = *(const float4*)(hW + (size_t)s * DD + c * 4);
  float4 b = *(const float4*)(rW + (size_t)e * DD + c * 4);
  float* p = agg + (size_t)d * DD + c * 4;
  atomicAdd(p + 0, a.x + b.x); atomicAdd(p + 1, a.y + b.y);
  atomicAdd(p + 2, a.z + b.z); atomicAdd(p + 3, a.w + b.w);
  if (c == 0) atomicAdd(deg + d, 1.0f);
}

// ---------------- Kernel C: finalize step ----------------
__global__ __launch_bounds__(WAVES * 32)
void node_finalize_kernel(const float* __restrict__ agg, const float* __restrict__ deg,
                          const float* __restrict__ hL, const float* __restrict__ ht,
                          const __bf16* __restrict__ WGhi, const __bf16* __restrict__ WGlo,
                          const float* __restrict__ gate_bias,
                          float* __restrict__ hout, int N) {
  __shared__ float xs_all[WAVES][ROWS_PER_WAVE][DD];
  __shared__ __align__(32) __bf16 bhbuf[2 * KTBLK];
  __shared__ __align__(32) __bf16 blbuf[2 * KTBLK];
  int tid = threadIdx.x;
  int lane = tid & 31;
  int wave = tid >> 5;
  float(*xs)[DD] = xs_all[wave];
  int row0 = blockIdx.x * ROWS_PER_BLOCK + wave * ROWS_PER_WAVE;
  int c0 = lane * 8;

  // Phase 1: cur = rrelu(agg/max(deg,1) + hL)
  for (int i = 0; i < 16; ++i) {
    int r = row0 + i;
    float dg = (r < N) ? fmaxf(deg[r], 1.0f) : 1.0f;
#pragma unroll
    for (int j = 0; j < 8; ++j) {
      float cv = 0.f;
      if (r < N) {
        cv = agg[(size_t)r * DD + c0 + j] / dg + hL[(size_t)r * DD + c0 + j];
        cv = (cv >= 0.f) ? cv : RRELU_SLOPE * cv;
      }
      xs[i][c0 + j] = cv;
    }
  }
  __syncthreads();

  // Phase 2: gate GEMM + mix (overwrite LDS with mix)
  {
    v8f acc[16];
#pragma unroll
    for (int nt = 0; nt < 16; ++nt) acc[nt] = (v8f){0, 0, 0, 0, 0, 0, 0, 0};
    gemm_slab_coop(&xs[0][0], WGhi, WGlo, bhbuf, blbuf, lane, tid, acc);
    int mbase = (lane & 16) ? 8 : 0;
    int colin = lane & 15;
#pragma unroll
    for (int nt = 0; nt < 16; ++nt) {
      int col = nt * 16 + colin;
      float gb = gate_bias[col];
#pragma unroll
      for (int v = 0; v < 8; ++v) {
        int rr = row0 + mbase + v;
        float cur = xs[mbase + v][col];
        float g = sigmoidf_(acc[nt][v] + gb);
        float hv = (rr < N) ? ht[(size_t)rr * DD + col] : 0.f;
        xs[mbase + v][col] = g * cur + (1.f - g) * hv;
      }
    }
  }
  __syncthreads();

  // Phase 3: expmap0 + store
  {
    float myscale = expmap0_scale(sqrtf(row_sumsq(xs, lane)));
    for (int i = 0; i < 16; ++i) {
      float s = __shfl(myscale, i, 32);
      int r = row0 + i;
      if (r < N) {
#pragma unroll
        for (int j = 0; j < 8; ++j)
          hout[(size_t)r * DD + c0 + j] = xs[i][c0 + j] * s;
      }
    }
  }
}

// ---------------- host launch ----------------
static inline size_t align_up(size_t x, size_t a) { return (x + a - 1) & ~(a - 1); }

extern "C" void kernel_launch(void* const* d_in, const int* in_sizes, int n_in,
                              void* d_out, int out_size, void* d_ws, size_t ws_size,
                              hipStream_t stream) {
  (void)n_in; (void)ws_size;
  const int* src   = (const int*)d_in[0];
  const int* dst   = (const int*)d_in[1];
  const int* etype = (const int*)d_in[2];
  const float* dynamic_emb = (const float*)d_in[3];
  const float* emb_rel     = (const float*)d_in[4];
  const float* time_weight = (const float*)d_in[5];
  const float* time_bias   = (const float*)d_in[6];
  const float* time_lin_w  = (const float*)d_in[7];
  const float* time_lin_b  = (const float*)d_in[8];
  const float* w_neighbor  = (const float*)d_in[9];
  const float* time_gate_b = (const float*)d_in[12];
  const float* gru_w_ih    = (const float*)d_in[13];
  const float* gru_w_hh    = (const float*)d_in[14];
  const float* gru_b_ih    = (const float*)d_in[15];
  const float* gru_b_hh    = (const float*)d_in[16];
  float* out = (float*)d_out;

  const int D = in_sizes[5];                  // 256
  const int N = in_sizes[3] / D;              // 40000
  const int R = in_sizes[4] / D;              // 500
  const int T = (out_size - R * D) / (N * D); // 8
  const long long E = (long long)in_sizes[0] / T;

  // workspace layout
  char* w = (char*)d_ws;
  size_t off = 0;
  auto take = [&](size_t bytes) { size_t o = off; off = align_up(off + bytes, 256); return (void*)(w + o); };
  float*  ht      = (float*)take((size_t)N * D * 4);
  float*  hW      = (float*)take((size_t)N * D * 4);
  float*  hL      = (float*)take((size_t)N * D * 4);
  float*  agg     = (float*)take((size_t)N * D * 4);
  float*  rel_sum = (float*)take((size_t)R * D * 4);
  float*  rel_cnt = (float*)take((size_t)R * 4);
  float*  deg     = (float*)take((size_t)N * 4);
  float*  r_emb   = (float*)take((size_t)R * D * 4);
  float*  rW      = (float*)take((size_t)R * D * 4);
  float*  tv      = (float*)take((size_t)D * 4);
  __bf16* Whi     = (__bf16*)take((size_t)4 * D * D * 2);
  __bf16* Wlo     = (__bf16*)take((size_t)4 * D * D * 2);

  const __bf16 *W1hi = Whi,             *W1lo = Wlo;
  const __bf16 *WNhi = Whi + 1 * D * D, *WNlo = Wlo + 1 * D * D;
  const __bf16 *WLhi = Whi + 2 * D * D, *WLlo = Wlo + 2 * D * D;
  const __bf16 *WGhi = Whi + 3 * D * D, *WGlo = Wlo + 3 * D * D;

  {
    int total = 4 * D * D;
    prep_weights_kernel<<<(total + 255) / 256, 256, 0, stream>>>(
        time_lin_w, w_neighbor, (const float*)d_in[10], (const float*)d_in[11], Whi, Wlo);
    hipMemcpyAsync(r_emb, emb_rel, (size_t)R * D * 4, hipMemcpyDeviceToDevice, stream);
  }

  const int nodeBlocks = (N + ROWS_PER_BLOCK - 1) / ROWS_PER_BLOCK;
  const long long edgeThreads = E * 64;
  const int edgeBlocks = (int)((edgeThreads + 255) / 256);

  for (int t = 0; t < T; ++t) {
    hipMemsetAsync(rel_sum, 0, (size_t)R * D * 4, stream);
    hipMemsetAsync(rel_cnt, 0, (size_t)R * 4, stream);
    hipMemsetAsync(agg,     0, (size_t)N * D * 4, stream);
    hipMemsetAsync(deg,     0, (size_t)N * 4, stream);

    float t_pos = (float)(T + 1 - t);
    time_vec_kernel<<<1, DD, 0, stream>>>(time_weight, time_bias, time_lin_w, time_lin_b, t_pos, tv);

    const float* hin = (t == 0) ? dynamic_emb : (out + (size_t)(t - 1) * N * D);
    node_update_kernel<<<nodeBlocks, WAVES * 32, 0, stream>>>(
        hin, tv, W1hi, W1lo, WNhi, WNlo, WLhi, WLlo, ht, hW, hL, N, (t == 0) ? 1 : 0);

    rel_scatter_kernel<<<edgeBlocks, 256, 0, stream>>>(
        src + (size_t)t * E, etype + (size_t)t * E, ht, rel_sum, rel_cnt, E);

    gru_kernel<<<R, 256, 0, stream>>>(emb_rel, rel_sum, rel_cnt,
                                      gru_w_ih, gru_w_hh, gru_b_ih, gru_b_hh,
                                      w_neighbor, r_emb, rW);

    agg_scatter_kernel<<<edgeBlocks, 256, 0, stream>>>(
        src + (size_t)t * E, dst + (size_t)t * E, etype + (size_t)t * E,
        hW, rW, agg, deg, E);

    node_finalize_kernel<<<nodeBlocks, WAVES * 32, 0, stream>>>(
        agg, deg, hL, ht, WGhi, WGlo, time_gate_b, out + (size_t)t * N * D, N);
  }

  hipMemcpyAsync(out + (size_t)T * N * D, r_emb, (size_t)R * D * 4,
                 hipMemcpyDeviceToDevice, stream);
}
